// DeltaNet_47382079209618
// MI455X (gfx1250) — compile-verified, matrix operand order checked by reading, not run on hardware
//
#include <hip/hip_runtime.h>
#include <cstdint>
#include <cstddef>

// ---------------------------------------------------------------- types
typedef __attribute__((ext_vector_type(16))) __bf16 v16bf;
typedef __attribute__((ext_vector_type(8)))  __bf16 v8bf;
typedef __attribute__((ext_vector_type(8)))  float  v8f;
typedef __attribute__((ext_vector_type(4)))  float  v4f;

static constexpr int Bc = 2, Lc = 4096, Dc = 2048, Hc = 16, DKc = 128;
static constexpr int CH = 64;               // delta-rule chunk length

// ---------------------------------------------------------------- helpers
__device__ inline __bf16 f2bf(float f) {
    union { float f; uint32_t u; } v; v.f = f;
    uint32_t r = v.u + 0x7FFFu + ((v.u >> 16) & 1u);
    uint16_t h = (uint16_t)(r >> 16);
    __bf16 b; __builtin_memcpy(&b, &h, 2); return b;
}
__device__ inline float bf2f(__bf16 b) {
    uint16_t h; __builtin_memcpy(&h, &b, 2);
    union { uint32_t u; float f; } v; v.u = ((uint32_t)h) << 16; return v.f;
}
__device__ inline v8f vzero() {
    v8f z;
    #pragma unroll
    for (int i = 0; i < 8; ++i) z[i] = 0.f;
    return z;
}

__device__ inline v8f wmma_bf16(v16bf a, v16bf b, v8f c) {
    return __builtin_amdgcn_wmma_f32_16x16x32_bf16(false, a, false, b,
                                                   (short)0, c, false, false);
}

// gfx1250 async global->LDS copy (16B per lane), tracked by ASYNCcnt.
// VDST = 32-bit LDS byte address (low 32 bits of the flat LDS address),
// VADDR = 64-bit global address (GV mode, saddr = off).
__device__ inline void async_ld_b128(void* lds, const void* g) {
    uint32_t l = (uint32_t)(uintptr_t)lds;
    uint64_t a = (uint64_t)(uintptr_t)g;
    asm volatile("global_load_async_to_lds_b128 %0, %1, off"
                 :: "v"(l), "v"(a) : "memory");
}
__device__ inline void wait_async0() {
    asm volatile("s_wait_asynccnt 0" ::: "memory");
}

// A-fragment (16x32 bf16) loader from LDS laid out [row][k] (row-major,
// stride in elements).  B-operands use the same loader on a [n][k] buffer.
// Layout per CDNA5 ISA: lanes 0-15 hold K 0-7 & 16-23, lanes 16-31 hold
// K 8-15 & 24-31, M/N = lane&15.
__device__ inline v16bf ldsfrag(const __bf16* p, int row0, int stride, int k0) {
    int l = threadIdx.x & 31;
    const __bf16* q = p + (size_t)(row0 + (l & 15)) * stride + k0 + ((l & 16) ? 8 : 0);
    v8bf lo = *(const v8bf*)(q);
    v8bf hi = *(const v8bf*)(q + 16);
    return __builtin_shufflevector(lo, hi, 0,1,2,3,4,5,6,7,8,9,10,11,12,13,14,15);
}

// ================================================================ GEMM
// C[M,N] = A[M,K] * B[K,N].  A: f32 (converted) or bf16 (async-copied),
// B: f32 weights, C: f32 or bf16.  Block tile 128x256, K-tile 32,
// 8 waves in a 2(M)x4(N) grid, each wave 64x64 = 4x4 wmma tiles.
template<bool A_F32, bool OUT_F32>
__global__ __launch_bounds__(256)
void gemm_wmma(const void* __restrict__ Ap, const float* __restrict__ Bp,
               void* __restrict__ Cp, int M, int N, int K)
{
    __shared__ __bf16 Al[128 * 32];   // [m][k]
    __shared__ __bf16 Bl[256 * 32];   // [n][k]  (B transposed)
    const int tid = threadIdx.x, lane = tid & 31, wave = tid >> 5;
    const int bm = blockIdx.y * 128, bn = blockIdx.x * 256;
    const int wm = (wave >> 2) * 64, wn = (wave & 3) * 64;

    v8f acc[4][4];
    #pragma unroll
    for (int i = 0; i < 4; ++i)
        #pragma unroll
        for (int j = 0; j < 4; ++j) acc[i][j] = vzero();

    for (int k0 = 0; k0 < K; k0 += 32) {
        // ---- load A tile 128x32
        if constexpr (A_F32) {
            const float* A = (const float*)Ap;
            #pragma unroll
            for (int q = 0; q < 4; ++q) {
                int e = tid * 16 + q * 4;
                int r = e >> 5, c = e & 31;
                v4f d = *(const v4f*)(A + (size_t)(bm + r) * K + k0 + c);
                __bf16* dst = Al + r * 32 + c;
                dst[0] = f2bf(d[0]); dst[1] = f2bf(d[1]);
                dst[2] = f2bf(d[2]); dst[3] = f2bf(d[3]);
            }
        } else {
            const __bf16* A = (const __bf16*)Ap;
            #pragma unroll
            for (int q = 0; q < 2; ++q) {
                int e = tid * 16 + q * 8;
                int r = e >> 5, c = e & 31;
                async_ld_b128(Al + r * 32 + c,
                              A + (size_t)(bm + r) * K + k0 + c);
            }
        }
        // ---- load B tile 32x256, store transposed [n][k]
        #pragma unroll
        for (int q = 0; q < 8; ++q) {
            int e = tid * 32 + q * 4;
            int kr = e >> 8, nc = e & 255;
            v4f d = *(const v4f*)(Bp + (size_t)(k0 + kr) * N + bn + nc);
            Bl[(nc + 0) * 32 + kr] = f2bf(d[0]);
            Bl[(nc + 1) * 32 + kr] = f2bf(d[1]);
            Bl[(nc + 2) * 32 + kr] = f2bf(d[2]);
            Bl[(nc + 3) * 32 + kr] = f2bf(d[3]);
        }
        // ---- prefetch next K-tile into caches
        if (k0 + 32 < K) {
            if constexpr (A_F32)
                __builtin_prefetch(((const float*)Ap) +
                    (size_t)(bm + (tid & 127)) * K + k0 + 32, 0, 1);
            else
                __builtin_prefetch(((const __bf16*)Ap) +
                    (size_t)(bm + (tid & 127)) * K + k0 + 32, 0, 1);
            __builtin_prefetch(Bp + (size_t)(k0 + 32 + (tid >> 3)) * N +
                               bn + (tid & 7) * 32, 0, 1);
        }
        if constexpr (!A_F32) wait_async0();
        __syncthreads();

        v16bf af[4], bfg[4];
        #pragma unroll
        for (int i = 0; i < 4; ++i) af[i]  = ldsfrag(Al, wm + i * 16, 32, 0);
        #pragma unroll
        for (int j = 0; j < 4; ++j) bfg[j] = ldsfrag(Bl, wn + j * 16, 32, 0);
        #pragma unroll
        for (int i = 0; i < 4; ++i)
            #pragma unroll
            for (int j = 0; j < 4; ++j)
                acc[i][j] = wmma_bf16(af[i], bfg[j], acc[i][j]);
        __syncthreads();
    }
    // ---- store
    #pragma unroll
    for (int i = 0; i < 4; ++i)
        #pragma unroll
        for (int j = 0; j < 4; ++j)
            #pragma unroll
            for (int r = 0; r < 8; ++r) {
                int row = bm + wm + i * 16 + (lane >> 4) * 8 + r;
                int col = bn + wn + j * 16 + (lane & 15);
                if constexpr (OUT_F32)
                    ((float*)Cp)[(size_t)row * N + col] = acc[i][j][r];
                else
                    ((__bf16*)Cp)[(size_t)row * N + col] = f2bf(acc[i][j][r]);
            }
}

// ================================================================ conv/norm/beta
// one block per (b,l): depthwise causal conv (K=4) + SiLU on q/k/v,
// per-head L2 norm of q,k; beta = sigmoid(hs @ w_beta).
__global__ __launch_bounds__(256)
void conv_kernel(const __bf16* __restrict__ qkv, const float* __restrict__ hs,
                 const float* __restrict__ wq, const float* __restrict__ wk,
                 const float* __restrict__ wv, const float* __restrict__ wbeta,
                 __bf16* __restrict__ qo, __bf16* __restrict__ ko,
                 __bf16* __restrict__ vo, float* __restrict__ betao)
{
    const int bl = blockIdx.x, b = bl / Lc, l = bl % Lc;
    const int tid = threadIdx.x;
    __shared__ float red[2][Hc];
    __shared__ float bred[Hc];
    if (tid < Hc) { red[0][tid] = 0.f; red[1][tid] = 0.f; bred[tid] = 0.f; }
    __syncthreads();

    const int c0 = tid * 8;
    float qv[8], kv[8], vv[8];
    #pragma unroll
    for (int x = 0; x < 8; ++x) { qv[x] = kv[x] = vv[x] = 0.f; }

    for (int i = 0; i < 4; ++i) {
        int ls = l - 3 + i;
        if (ls < 0) continue;
        size_t rb = ((size_t)(b * Lc + ls)) * (3 * Dc);
        #pragma unroll
        for (int x = 0; x < 8; ++x) {
            int c = c0 + x;
            qv[x] += bf2f(qkv[rb + c])           * wq[c * 4 + i];
            kv[x] += bf2f(qkv[rb + Dc + c])      * wk[c * 4 + i];
            vv[x] += bf2f(qkv[rb + 2 * Dc + c])  * wv[c * 4 + i];
        }
    }
    float sq = 0.f, sk = 0.f;
    #pragma unroll
    for (int x = 0; x < 8; ++x) {
        qv[x] = qv[x] / (1.f + __expf(-qv[x]));
        kv[x] = kv[x] / (1.f + __expf(-kv[x]));
        vv[x] = vv[x] / (1.f + __expf(-vv[x]));
        sq += qv[x] * qv[x];
        sk += kv[x] * kv[x];
    }
    const int head = c0 >> 7;
    atomicAdd(&red[0][head], sq);
    atomicAdd(&red[1][head], sk);

    // beta partials
    float bacc[Hc];
    #pragma unroll
    for (int h = 0; h < Hc; ++h) bacc[h] = 0.f;
    size_t hb = (size_t)bl * Dc;
    #pragma unroll
    for (int x = 0; x < 8; ++x) {
        float hv = hs[hb + c0 + x];
        const float* wb = wbeta + (size_t)(c0 + x) * Hc;
        #pragma unroll
        for (int h = 0; h < Hc; ++h) bacc[h] += hv * wb[h];
    }
    #pragma unroll
    for (int h = 0; h < Hc; ++h) atomicAdd(&bred[h], bacc[h]);
    __syncthreads();

    float rq = rsqrtf(red[0][head] + 1e-12f);
    float rk = rsqrtf(red[1][head] + 1e-12f);
    size_t ob = (((size_t)(b * Hc + head)) * Lc + l) * DKc + (c0 & 127);
    #pragma unroll
    for (int x = 0; x < 8; ++x) {
        qo[ob + x] = f2bf(qv[x] * rq);
        ko[ob + x] = f2bf(kv[x] * rk);
        vo[ob + x] = f2bf(vv[x]);
    }
    if (tid < Hc) {
        float s = bred[tid];
        betao[((size_t)(b * Hc + tid)) * Lc + l] = 1.f / (1.f + __expf(-s));
    }
}

// ================================================================ delta rule
// chunked delta rule, one workgroup per (b,h); 8 waves; state S[128][128]
// lives in WMMA accumulators (wave w owns dk rows [16w,16w+16), tiles j=0..7).
static constexpr int SM_ST   = 0;        // S^T bf16  [128][128]   32768
static constexpr int SM_KC   = 32768;    // K chunk   [64][128]    16384
static constexpr int SM_KCT  = 49152;    // K^T       [128][64]    16384
static constexpr int SM_QC   = 65536;    // Q chunk   [64][128]    16384
static constexpr int SM_VC   = 81920;    // V chunk   [64][128]    16384
static constexpr int SM_UF   = 98304;    // U f32     [64][128]    32768
static constexpr int SM_UT   = 131072;   // U^T bf16  [128][64]    16384
static constexpr int SM_AATT = 147456;   // A f32     [64][64]     16384
static constexpr int SM_WQK  = 163840;   // mask(QK^T)[64][64]      8192
static constexpr int SM_OL   = 172032;   // O f32     [64][128]    32768
static constexpr int SM_BETA = 204800;   // beta f32  [64]           256
static constexpr int SMEM_BYTES = 205056;

__global__ __launch_bounds__(256)
void delta_kernel(const __bf16* __restrict__ qg, const __bf16* __restrict__ kg,
                  const __bf16* __restrict__ vg, const float* __restrict__ betag,
                  const float* __restrict__ onormw, __bf16* __restrict__ onorm)
{
    extern __shared__ __align__(16) char smem[];
    __bf16* ST   = (__bf16*)(smem + SM_ST);
    __bf16* Kc   = (__bf16*)(smem + SM_KC);
    __bf16* KcT  = (__bf16*)(smem + SM_KCT);
    __bf16* Qc   = (__bf16*)(smem + SM_QC);
    __bf16* Vc   = (__bf16*)(smem + SM_VC);
    float*  Uf   = (float*)(smem + SM_UF);
    __bf16* Ut   = (__bf16*)(smem + SM_UT);
    float*  Aatt = (float*)(smem + SM_AATT);
    __bf16* Wqk  = (__bf16*)(smem + SM_WQK);
    float*  Ol   = (float*)(smem + SM_OL);
    float*  betac= (float*)(smem + SM_BETA);

    const int bh = blockIdx.x, b = bh >> 4, h = bh & 15;
    const int tid = threadIdx.x, lane = tid & 31, wave = tid >> 5;
    const __bf16* qbase = qg + (size_t)bh * Lc * DKc;
    const __bf16* kbase = kg + (size_t)bh * Lc * DKc;
    const __bf16* vbase = vg + (size_t)bh * Lc * DKc;
    const float*  bbase = betag + (size_t)bh * Lc;

    const int solve_j = tid & 127, solve_p = tid >> 7;

    v8f Sacc[8];
    #pragma unroll
    for (int j = 0; j < 8; ++j) Sacc[j] = vzero();

    for (int ch = 0; ch < Lc / CH; ++ch) {
        const int l0 = ch * CH;
        // ---- phase 1: async copy Q/V chunk to LDS; K manually (+K^T);
        //      beta; spill S^T (bf16)
        for (int e = tid; e < CH * DKc / 8; e += 256) {
            int t = e >> 4, d0 = (e & 15) << 3;
            size_t gofs = (size_t)(l0 + t) * DKc + d0;
            async_ld_b128(Qc + t * DKc + d0, qbase + gofs);
            async_ld_b128(Vc + t * DKc + d0, vbase + gofs);
            v8bf kvv = *(const v8bf*)(kbase + gofs);
            *(v8bf*)(Kc + t * DKc + d0) = kvv;
            #pragma unroll
            for (int x = 0; x < 8; ++x) KcT[(d0 + x) * CH + t] = kvv[x];
        }
        if (tid < CH) betac[tid] = bbase[l0 + tid];
        #pragma unroll
        for (int j = 0; j < 8; ++j)
            #pragma unroll
            for (int r = 0; r < 8; ++r) {
                int dk = wave * 16 + (lane >> 4) * 8 + r;
                int dv = j * 16 + (lane & 15);
                ST[dv * DKc + dk] = f2bf(Sacc[j][r]);
            }
        wait_async0();
        __syncthreads();

        // ---- phase 2: A = beta*tril(KK^T,-1); W = tril(QK^T)  (16 tiles)
        #pragma unroll
        for (int q = 0; q < 2; ++q) {
            int tt = wave * 2 + q;
            int it = (tt >> 2) * 16, is = (tt & 3) * 16;
            v8f ak = vzero(), aq = vzero();
            #pragma unroll
            for (int k0 = 0; k0 < DKc; k0 += 32) {
                v16bf bf = ldsfrag(Kc, is, DKc, k0);
                ak = wmma_bf16(ldsfrag(Kc, it, DKc, k0), bf, ak);
                aq = wmma_bf16(ldsfrag(Qc, it, DKc, k0), bf, aq);
            }
            #pragma unroll
            for (int r = 0; r < 8; ++r) {
                int t = it + (lane >> 4) * 8 + r;
                int s = is + (lane & 15);
                float bt = betac[t];                 // unconditional LDS read
                float av = bt * ak[r];
                Aatt[t * CH + s] = (s < t) ? av : 0.f;
                Wqk[t * CH + s]  = f2bf((s <= t) ? aq[r] : 0.f);
            }
        }
        // ---- phase 3: X = beta*(V - K@S_prev)  (32 tiles)
        #pragma unroll
        for (int q = 0; q < 4; ++q) {
            int tt = wave * 4 + q;
            int it = (tt >> 3) * 16, jv = (tt & 7) * 16;
            v8f x = vzero();
            #pragma unroll
            for (int k0 = 0; k0 < DKc; k0 += 32)
                x = wmma_bf16(ldsfrag(Kc, it, DKc, k0),
                              ldsfrag(ST, jv, DKc, k0), x);
            #pragma unroll
            for (int r = 0; r < 8; ++r) {
                int t = it + (lane >> 4) * 8 + r;
                int j = jv + (lane & 15);
                float bt = betac[t];
                Uf[t * DKc + j] = bt * (bf2f(Vc[t * DKc + j]) - x[r]);
            }
        }
        __syncthreads();

        // ---- phase 4: forward-substitution: u_t -= sum_{s<t} A[t,s] u_s
        for (int s = 0; s < CH - 1; ++s) {
            float us = Uf[s * DKc + solve_j];
            for (int t = s + 1 + solve_p; t < CH; t += 2)
                Uf[t * DKc + solve_j] -= Aatt[t * CH + s] * us;
            __syncthreads();
        }
        // ---- phase 5: U^T (bf16)
        for (int e = tid; e < CH * DKc; e += 256) {
            int t = e >> 7, j = e & 127;
            Ut[j * CH + t] = f2bf(Uf[e]);
        }
        __syncthreads();

        // ---- phase 6: O = Q@S_prev + tril(QK^T)@U  (32 tiles)
        #pragma unroll
        for (int q = 0; q < 4; ++q) {
            int tt = wave * 4 + q;
            int it = (tt >> 3) * 16, jv = (tt & 7) * 16;
            v8f o = vzero();
            #pragma unroll
            for (int k0 = 0; k0 < DKc; k0 += 32)
                o = wmma_bf16(ldsfrag(Qc, it, DKc, k0),
                              ldsfrag(ST, jv, DKc, k0), o);
            #pragma unroll
            for (int k0 = 0; k0 < CH; k0 += 32)
                o = wmma_bf16(ldsfrag(Wqk, it, CH, k0),
                              ldsfrag(Ut, jv, CH, k0), o);
            #pragma unroll
            for (int r = 0; r < 8; ++r) {
                int t = it + (lane >> 4) * 8 + r;
                int j = jv + (lane & 15);
                Ol[t * DKc + j] = o[r];
            }
        }
        // ---- phase 7: S += K^T @ U  (accumulators in place)
        #pragma unroll
        for (int j = 0; j < 8; ++j)
            #pragma unroll
            for (int k0 = 0; k0 < CH; k0 += 32)
                Sacc[j] = wmma_bf16(ldsfrag(KcT, wave * 16, CH, k0),
                                    ldsfrag(Ut, j * 16, CH, k0), Sacc[j]);
        __syncthreads();

        // ---- phase 8: RMSNorm rows of O, store bf16 [b,l,h*128+dk]
        if (tid < CH) {
            int t = tid;
            float ss = 0.f;
            for (int j = 0; j < DKc; ++j) { float o = Ol[t * DKc + j]; ss += o * o; }
            float rs = rsqrtf(ss / (float)DKc + 1e-5f);
            size_t base = ((size_t)(b * Lc + l0 + t)) * Dc + (size_t)h * DKc;
            for (int j = 0; j < DKc; ++j)
                onorm[base + j] = f2bf(Ol[t * DKc + j] * rs * onormw[j]);
        }
        __syncthreads();
    }
}

// ================================================================ host
static constexpr size_t WS_QKV  = 0;           // bf16 [B*L][3D]   100663296 B
static constexpr size_t WS_Q    = 100663296;   // bf16 [B,H,L,DK]   33554432 B
static constexpr size_t WS_K    = 134217728;
static constexpr size_t WS_V    = 167772160;
static constexpr size_t WS_BETA = 201326592;   // f32  [B,H,L]        524288 B
static constexpr size_t WS_ON   = 201850880;   // bf16 [B*L][D]     33554432 B
// total required: 235405312 bytes

extern "C" void kernel_launch(void* const* d_in, const int* in_sizes, int n_in,
                              void* d_out, int out_size, void* d_ws, size_t ws_size,
                              hipStream_t stream)
{
    const float* hs      = (const float*)d_in[0];
    const float* w_cattn = (const float*)d_in[1];
    const float* wq_conv = (const float*)d_in[2];
    const float* wk_conv = (const float*)d_in[3];
    const float* wv_conv = (const float*)d_in[4];
    const float* w_beta  = (const float*)d_in[5];
    const float* o_norm  = (const float*)d_in[6];
    const float* w_o     = (const float*)d_in[7];

    char* ws = (char*)d_ws;
    __bf16* qkv   = (__bf16*)(ws + WS_QKV);
    __bf16* qb    = (__bf16*)(ws + WS_Q);
    __bf16* kb    = (__bf16*)(ws + WS_K);
    __bf16* vb    = (__bf16*)(ws + WS_V);
    float*  betab = (float*)(ws + WS_BETA);
    __bf16* onb   = (__bf16*)(ws + WS_ON);

    const int M = Bc * Lc;           // 8192
    dim3 blk(256);

    // 1) qkv projection:  [8192,2048] x [2048,6144] -> bf16
    gemm_wmma<true, false><<<dim3((3 * Dc) / 256, M / 128), blk, 0, stream>>>(
        hs, w_cattn, qkv, M, 3 * Dc, Dc);

    // 2) conv + silu + l2norm + beta
    conv_kernel<<<dim3(M), blk, 0, stream>>>(
        qkv, hs, wq_conv, wk_conv, wv_conv, w_beta, qb, kb, vb, betab);

    // 3) chunked delta rule (state in WMMA accumulators), fused RMSNorm
    (void)hipFuncSetAttribute((const void*)delta_kernel,
                              hipFuncAttributeMaxDynamicSharedMemorySize, SMEM_BYTES);
    delta_kernel<<<dim3(Bc * Hc), blk, SMEM_BYTES, stream>>>(
        qb, kb, vb, betab, o_norm, onb);

    // 4) output projection: [8192,2048] x [2048,2048] -> f32 out
    gemm_wmma<false, true><<<dim3(Dc / 256, M / 128), blk, 0, stream>>>(
        onb, w_o, (float*)d_out, M, Dc, Dc);
}
